// DecoderLayer_833223656504
// MI455X (gfx1250) — matrix operand order repaired and checked by reading, not compiled, not run on hardware
//
#include <hip/hip_runtime.h>

// ---------------------------------------------------------------------------
// Problem constants (match reference: B=2, S=2048, D=1024, H=16, DH=64, F=4096)
// ---------------------------------------------------------------------------
constexpr int Bq  = 2;
constexpr int Sq  = 2048;
constexpr int Dq  = 1024;
constexpr int Hq  = 16;
constexpr int DHq = 64;
constexpr int Fq  = 4096;

typedef __attribute__((ext_vector_type(16))) _Float16 v16h;
typedef __attribute__((ext_vector_type(8)))  _Float16 v8h;
typedef __attribute__((ext_vector_type(8)))  float    v8f;
typedef __attribute__((ext_vector_type(4)))  float    v4f;

union F16V { v16h v; v8h p[2]; _Float16 e[16]; };

// ---------------------------------------------------------------------------
// CDNA5 async global->LDS copy path (ASYNCcnt + s_wait_asynccnt), with a
// synchronous fallback if the toolchain does not expose the builtins.
// Builtin prototype (from hipcc diagnostics): the b128 variant takes pointers
// to 128-bit int vectors — global (AS1) source, LDS (AS3) destination.
// ---------------------------------------------------------------------------
#if __has_builtin(__builtin_amdgcn_global_load_async_to_lds_b128) && \
    __has_builtin(__builtin_amdgcn_s_wait_asynccnt)
#define USE_ASYNC 1
#else
#define USE_ASYNC 0
#endif

#if USE_ASYNC
typedef int v4i __attribute__((vector_size(16)));
typedef __attribute__((address_space(1))) v4i* gb128_t;  // global 128b chunk
typedef __attribute__((address_space(3))) v4i* lb128_t;  // LDS    128b chunk
// copy 16 f16 (32B) global -> LDS as two async B128 transfers
#define CP16(g, l)                                                             \
  do {                                                                         \
    __builtin_amdgcn_global_load_async_to_lds_b128((gb128_t)(g),               \
                                                   (lb128_t)(l), 0, 0);        \
    __builtin_amdgcn_global_load_async_to_lds_b128((gb128_t)((g) + 8),         \
                                                   (lb128_t)((l) + 8), 0, 0);  \
  } while (0)
#define ASYNC_WAIT() __builtin_amdgcn_s_wait_asynccnt(0)
#else
#define CP16(g, l) (*(v16h*)(l) = *(const v16h*)(g))
#define ASYNC_WAIT() ((void)0)
#endif

// D = A(16x32 f16) * B(32x16 f16) + C(16x16 f32), wave32 WMMA
__device__ __forceinline__ v8f wmma_f16(v16h a, v16h b, v8f c) {
  return __builtin_amdgcn_wmma_f32_16x16x32_f16(false, a, false, b, (short)0, c,
                                                false, false);
}

// A-fragment: lane holds row M=lane&15; K chunks [8h,8h+8) and [16+8h,16+8h+8)
__device__ __forceinline__ v16h frag_a(const _Float16* row, int half) {
  F16V u;
  u.p[0] = *(const v8h*)(row + 8 * half);
  u.p[1] = *(const v8h*)(row + 16 + 8 * half);
  return u.v;
}
// B-fragment: lane holds column N=lane&15; K chunks [8h,8h+8) and [16+8h,16+8h+8)
__device__ __forceinline__ v16h frag_b(const _Float16* row, int half) {
  F16V u;
  u.p[0] = *(const v8h*)(row + 8 * half);
  u.p[1] = *(const v8h*)(row + 16 + 8 * half);
  return u.v;
}

// ---------------------------------------------------------------------------
// f32 -> f16 conversions
// ---------------------------------------------------------------------------
__global__ void cvt_f16_kernel(const float* __restrict__ in,
                               _Float16* __restrict__ out, long n) {
  long i = (long)blockIdx.x * 256 + threadIdx.x;
  if (i < n) out[i] = (_Float16)in[i];
}

// W_O[d][dh][h] -> WOt[d][h*64+dh] so K index matches z2's (h,dh) flattening
__global__ void cvt_wot_kernel(const float* __restrict__ in,
                               _Float16* __restrict__ out) {
  long i = (long)blockIdx.x * 256 + threadIdx.x;
  const long n = (long)Dq * DHq * Hq;
  if (i >= n) return;
  int  hh = (int)(i % Hq);
  long t  = i / Hq;
  int  dh = (int)(t % DHq);
  long d  = t / DHq;
  out[d * (long)(Hq * DHq) + hh * DHq + dh] = (_Float16)in[i];
}

// ---------------------------------------------------------------------------
// LayerNorm (f32 stats, f16 output): one row of D=1024 per 256-thread block
// ---------------------------------------------------------------------------
__global__ __launch_bounds__(256) void ln_kernel(const float* __restrict__ x,
                                                 const float* __restrict__ w,
                                                 const float* __restrict__ bb,
                                                 _Float16* __restrict__ y) {
  const int D = Dq;
  __shared__ float sbuf[16];
  const long  row = blockIdx.x;
  const float* xr = x + row * D;
  const int tid = threadIdx.x;
  float s = 0.f, s2 = 0.f;
  for (int i = tid; i < D; i += 256) { float v = xr[i]; s += v; s2 += v * v; }
  for (int o = 16; o; o >>= 1) {
    s  += __shfl_xor(s, o, 32);
    s2 += __shfl_xor(s2, o, 32);
  }
  if ((tid & 31) == 0) { sbuf[tid >> 5] = s; sbuf[8 + (tid >> 5)] = s2; }
  __syncthreads();
  s = 0.f; s2 = 0.f;
  for (int i = 0; i < 8; ++i) { s += sbuf[i]; s2 += sbuf[8 + i]; }
  float mu  = s / D;
  float var = s2 / D - mu * mu;
  float inv = rsqrtf(var + 1e-5f);
  for (int i = tid; i < D; i += 256)
    y[row * D + i] = (_Float16)((xr[i] - mu) * inv * w[i] + bb[i]);
}

// ---------------------------------------------------------------------------
// Generic NT GEMM: C = A(MxK f16, row-major) * B(NxK f16, row-major)^T
// 128x64 block tile, 4 waves (2Mx2N), each wave 64x32 via 4x2 WMMA accums.
// Double-buffered LDS; tiles staged with async global->LDS copies (ASYNCcnt)
// so the next K-slab streams in while the current one feeds the WMMAs.
// flags: 1=ReLU, 2=f16 output, 4=f16 transposed output (C[n*ldc+m])
// batch z: A offset (z/aDiv)*aStride, B offset (z%bMod)*bStride, C z*cStride.
// ---------------------------------------------------------------------------
__global__ __launch_bounds__(128) void gemm_nt_kernel(
    const _Float16* __restrict__ A, long aStride, int aDiv, int lda,
    const _Float16* __restrict__ B, long bStride, int bMod, int ldb,
    void* __restrict__ Cv, long cStride, int ldc,
    const float* __restrict__ bias, const float* __restrict__ residual,
    int K, int flags) {
  // tile buffer: A rows [0,128) then B rows [128,192), 32 halves per row
  __shared__ __align__(32) _Float16 tile[2][(128 + 64) * 32];
  const int tid = threadIdx.x, lane = tid & 31, wave = tid >> 5;
  const int wm = wave >> 1, wn = wave & 1;
  const int z = blockIdx.z;
  const int m0 = blockIdx.y * 128, n0 = blockIdx.x * 64;
  const _Float16* Ab = A + (long)(z / aDiv) * aStride + (long)m0 * lda;
  const _Float16* Bb = B + (long)(z % bMod) * bStride + (long)n0 * ldb;
  const int half = lane >> 4, nl = lane & 15;

  auto stage = [&](int k0, int bufi) {
    #pragma unroll
    for (int j = 0; j < 3; ++j) {               // 384 chunks of 16 halves
      int c = tid + 128 * j;
      int row = c >> 1, col = (c & 1) * 16;
      const _Float16* g = (row < 128)
          ? Ab + (long)row * lda + (k0 + col)
          : Bb + (long)(row - 128) * ldb + (k0 + col);
      CP16(g, &tile[bufi][row * 32 + col]);
    }
  };

  v8f acc[4][2] = {};
  const int nk = K / 32;
  stage(0, 0);
  for (int kk = 0; kk < nk; ++kk) {
    const int bufi = kk & 1;
    ASYNC_WAIT();                 // this wave's pending tile is in LDS
    __syncthreads();              // whole tile present; prev compute retired
    if (kk + 1 < nk) stage((kk + 1) * 32, bufi ^ 1);
    const _Float16* As = &tile[bufi][0];
    const _Float16* Bs = &tile[bufi][128 * 32];
    v16h b0 = frag_b(Bs + (wn * 32 + nl) * 32, half);
    v16h b1 = frag_b(Bs + (wn * 32 + 16 + nl) * 32, half);
    #pragma unroll
    for (int mi = 0; mi < 4; ++mi) {
      v16h a = frag_a(As + (wm * 64 + mi * 16 + nl) * 32, half);
      acc[mi][0] = wmma_f16(a, b0, acc[mi][0]);
      acc[mi][1] = wmma_f16(a, b1, acc[mi][1]);
    }
  }
  // Epilogue. C layout: lanes 0-15 -> M=r, N=lane; lanes 16-31 -> M=8+r.
  #pragma unroll
  for (int mi = 0; mi < 4; ++mi)
    #pragma unroll
    for (int ni = 0; ni < 2; ++ni) {
      int n = n0 + wn * 32 + ni * 16 + nl;
      float bv = bias ? bias[n] : 0.0f;
      #pragma unroll
      for (int r = 0; r < 8; ++r) {
        int m = m0 + wm * 64 + mi * 16 + half * 8 + r;
        float v = acc[mi][ni][r] + bv;
        if (flags & 1) v = fmaxf(v, 0.0f);
        if (residual) v += residual[(long)m * ldc + n];
        if (flags & 4) {
          ((_Float16*)Cv)[(long)z * cStride + (long)n * ldc + m] = (_Float16)v;
        } else if (flags & 2) {
          ((_Float16*)Cv)[(long)z * cStride + (long)m * ldc + n] = (_Float16)v;
        } else {
          ((float*)Cv)[(long)z * cStride + (long)m * ldc + n] = v;
        }
      }
    }
}

// ---------------------------------------------------------------------------
// Fused causal attention per (b,h, q-tile of 64). 4 waves / block.
// Phase 1: logits = (Q K^T)/8 with causal mask -> att buffer (raw), online
//          row max/sum via shfl reductions matched to the WMMA C layout.
// Phase 1.5: normalize att in place (this IS the required att f32 output).
// Phase 2: z = att * V^T via WMMA (att re-read as f16 A-fragments, Vt rows
//          are contiguous over p so B-fragments are plain 32B loads).
// ---------------------------------------------------------------------------
__global__ __launch_bounds__(128) void attn_kernel(
    const _Float16* __restrict__ Q, const _Float16* __restrict__ Km,
    const _Float16* __restrict__ Vt, float* __restrict__ att,
    _Float16* __restrict__ z2) {
  const int S = Sq, H = Hq;
  __shared__ __align__(32) _Float16 Qs[64][64];
  __shared__ __align__(32) _Float16 Ks[64][64];
  __shared__ float rowM[64], rowL[64];
  __shared__ float pMax[2][64], pSum[2][64];
  const int bh = blockIdx.y, b = bh / H, h = bh % H;
  const int q0 = blockIdx.x * 64;
  const _Float16* Qb = Q + (long)bh * S * DHq;
  const _Float16* Kb = Km + (long)bh * S * DHq;
  const _Float16* Vb = Vt + (long)bh * DHq * S;  // [dh][p]
  float* attb = att + (long)bh * S * S;
  const int tid = threadIdx.x, lane = tid & 31, wave = tid >> 5;
  const int wq = wave >> 1, wlo = wave & 1;  // phase1: p-half; phase2: dh-half
  const int half = lane >> 4, nl = lane & 15;

  {  // stage Q tile: 64 rows x 64 halves (async; waited before first use)
    int row = tid >> 1, c = (tid & 1) * 32;
    const _Float16* src = Qb + (long)(q0 + row) * DHq + c;
    CP16(src, &Qs[row][c]);
    CP16(src + 16, &Qs[row][c + 16]);
  }
  if (tid < 64) { rowM[tid] = -1e30f; rowL[tid] = 0.0f; }

  const int pEnd = q0 + 64;  // causal: only p-tiles up to the q-tile diagonal

  // ---------------- Phase 1 ----------------
  for (int p0 = 0; p0 < pEnd; p0 += 64) {
    __syncthreads();  // previous Ks consumers finished
    {
      int row = tid >> 1, c = (tid & 1) * 32;
      const _Float16* src = Kb + (long)(p0 + row) * DHq + c;
      CP16(src, &Ks[row][c]);
      CP16(src + 16, &Ks[row][c + 16]);
    }
    ASYNC_WAIT();
    __syncthreads();
    v8f acc[2][2] = {};
    #pragma unroll
    for (int ks = 0; ks < 2; ++ks) {  // dh = 64 -> two K-steps of 32
      v16h a0 = frag_a(&Qs[wq * 32 + nl][ks * 32], half);
      v16h a1 = frag_a(&Qs[wq * 32 + 16 + nl][ks * 32], half);
      v16h b0 = frag_b(&Ks[wlo * 32 + nl][ks * 32], half);
      v16h b1 = frag_b(&Ks[wlo * 32 + 16 + nl][ks * 32], half);
      acc[0][0] = wmma_f16(a0, b0, acc[0][0]);
      acc[0][1] = wmma_f16(a0, b1, acc[0][1]);
      acc[1][0] = wmma_f16(a1, b0, acc[1][0]);
      acc[1][1] = wmma_f16(a1, b1, acc[1][1]);
    }
    #pragma unroll
    for (int mi = 0; mi < 2; ++mi) {
      #pragma unroll
      for (int r = 0; r < 8; ++r) {
        int mloc = wq * 32 + mi * 16 + half * 8 + r;
        int m = q0 + mloc;
        int n0g = p0 + wlo * 32 + nl;
        int n1g = n0g + 16;
        float v0 = acc[mi][0][r] * 0.125f;  // 1/sqrt(64)
        float v1 = acc[mi][1][r] * 0.125f;
        if (n0g > m) v0 = -1e30f;
        if (n1g > m) v1 = -1e30f;
        attb[(long)m * S + n0g] = v0;
        attb[(long)m * S + n1g] = v1;
        // row stats: reduce across the 16-lane group holding this row
        float mx = fmaxf(v0, v1);
        mx = fmaxf(mx, __shfl_xor(mx, 1, 32));
        mx = fmaxf(mx, __shfl_xor(mx, 2, 32));
        mx = fmaxf(mx, __shfl_xor(mx, 4, 32));
        mx = fmaxf(mx, __shfl_xor(mx, 8, 32));
        float sm = __expf(v0 - mx) + __expf(v1 - mx);
        sm += __shfl_xor(sm, 1, 32);
        sm += __shfl_xor(sm, 2, 32);
        sm += __shfl_xor(sm, 4, 32);
        sm += __shfl_xor(sm, 8, 32);
        if (nl == 0) { pMax[wlo][mloc] = mx; pSum[wlo][mloc] = sm; }
      }
    }
    __syncthreads();
    if (tid < 64) {  // online merge of the tile's (m,l) into running stats
      float m0p = pMax[0][tid], m1p = pMax[1][tid];
      float tm = fmaxf(m0p, m1p);
      float tl = pSum[0][tid] * __expf(m0p - tm) + pSum[1][tid] * __expf(m1p - tm);
      float mo = rowM[tid];
      float mn = fmaxf(mo, tm);
      rowL[tid] = rowL[tid] * __expf(mo - mn) + tl * __expf(tm - mn);
      rowM[tid] = mn;
    }
  }
  __threadfence();
  __syncthreads();

  // ---------------- Phase 1.5: normalize att in place (final output) -------
  for (int rr = wave; rr < 64; rr += 4) {
    float mrow = rowM[rr];
    float inv  = 1.0f / rowL[rr];
    float* rowp = attb + (long)(q0 + rr) * S;
    for (int p = lane * 4; p < pEnd; p += 128) {
      v4f xv = *(const v4f*)(rowp + p);
      v4f ov;
      #pragma unroll
      for (int j = 0; j < 4; ++j) ov[j] = __expf(xv[j] - mrow) * inv;
      *(v4f*)(rowp + p) = ov;
    }
  }
  __threadfence();
  __syncthreads();

  // ---------------- Phase 2: z = att * V^T ----------------
  v8f zacc[2][2] = {};
  for (int p0 = 0; p0 < pEnd; p0 += 64) {
    if (p0 + 64 < pEnd) {  // prefetch next p-slab of V^T (global_prefetch)
      __builtin_prefetch(Vb + (long)(wlo * 32 + nl) * S + (p0 + 64), 0, 0);
    }
    #pragma unroll
    for (int ks = 0; ks < 2; ++ks) {
      int pb = p0 + ks * 32;
      v16h afr[2];
      #pragma unroll
      for (int mi = 0; mi < 2; ++mi) {
        const float* src = attb + (long)(q0 + wq * 32 + mi * 16 + nl) * S + pb;
        F16V u;
        #pragma unroll
        for (int j = 0; j < 8; ++j) u.e[j]     = (_Float16)src[8 * half + j];
        #pragma unroll
        for (int j = 0; j < 8; ++j) u.e[8 + j] = (_Float16)src[16 + 8 * half + j];
        afr[mi] = u.v;
      }
      v16h bfr[2];
      #pragma unroll
      for (int ni = 0; ni < 2; ++ni) {
        int n = wlo * 32 + ni * 16 + nl;  // head-dim column; Vt row contiguous
        bfr[ni] = frag_b(Vb + (long)n * S + pb, half);
      }
      zacc[0][0] = wmma_f16(afr[0], bfr[0], zacc[0][0]);
      zacc[0][1] = wmma_f16(afr[0], bfr[1], zacc[0][1]);
      zacc[1][0] = wmma_f16(afr[1], bfr[0], zacc[1][0]);
      zacc[1][1] = wmma_f16(afr[1], bfr[1], zacc[1][1]);
    }
  }
  // store z into z2[b, q, h*64 + dh] (flattened for the W_O NT GEMM)
  #pragma unroll
  for (int mi = 0; mi < 2; ++mi)
    #pragma unroll
    for (int ni = 0; ni < 2; ++ni) {
      int col = h * 64 + wlo * 32 + ni * 16 + nl;
      #pragma unroll
      for (int r = 0; r < 8; ++r) {
        int q = q0 + wq * 32 + mi * 16 + half * 8 + r;
        z2[((long)b * S + q) * Dq + col] = (_Float16)zacc[mi][ni][r];
      }
    }
}

// Zero the never-visited strictly-upper tiles of att (p >= q_tile_end).
__global__ __launch_bounds__(256) void zero_att_kernel(float* __restrict__ att) {
  const int S = Sq;
  int q = blockIdx.x;
  long bh = blockIdx.y;
  int p0 = ((q >> 6) + 1) << 6;
  float* row = att + (bh * S + q) * (long)S;
  for (int p = p0 + (int)threadIdx.x * 4; p < S; p += 1024) {
    v4f zz = {0.f, 0.f, 0.f, 0.f};
    *(v4f*)(row + p) = zz;
  }
}

// ---------------------------------------------------------------------------
// Host-side orchestration
// ---------------------------------------------------------------------------
extern "C" void kernel_launch(void* const* d_in, const int* in_sizes, int n_in,
                              void* d_out, int out_size, void* d_ws,
                              size_t ws_size, hipStream_t stream) {
  (void)in_sizes; (void)n_in; (void)out_size; (void)ws_size;
  const float* x    = (const float*)d_in[0];
  const float* W_Q  = (const float*)d_in[1];
  const float* W_K  = (const float*)d_in[2];
  const float* W_V  = (const float*)d_in[3];
  const float* W_O  = (const float*)d_in[4];
  const float* lnaw = (const float*)d_in[5];
  const float* lnab = (const float*)d_in[6];
  const float* lnmw = (const float*)d_in[7];
  const float* lnmb = (const float*)d_in[8];
  const float* w1   = (const float*)d_in[9];
  const float* b1   = (const float*)d_in[10];
  const float* w2   = (const float*)d_in[11];
  const float* b2   = (const float*)d_in[12];

  char* ws = (char*)d_ws;
  const long MB = 1 << 20;
  _Float16* yln  = (_Float16*)(ws + 0);        // 8MB  [B*S*D]   (dead after QKV)
  _Float16* Qh   = (_Float16*)(ws + 8 * MB);   // 8MB  [B,H,S,DH]
  _Float16* Kh   = (_Float16*)(ws + 16 * MB);  // 8MB
  _Float16* Vth  = (_Float16*)(ws + 24 * MB);  // 8MB  [B,H,DH,S] (transposed)
  _Float16* h16  = (_Float16*)(ws + 0);        // 32MB [B*S,F] overlays dead bufs
  _Float16* Wq16 = (_Float16*)(ws + 32 * MB);  // 2MB
  _Float16* Wk16 = (_Float16*)(ws + 34 * MB);  // 2MB
  _Float16* Wv16 = (_Float16*)(ws + 36 * MB);  // 2MB
  _Float16* WOt  = (_Float16*)(ws + 38 * MB);  // 2MB  [D, h*64+dh]
  _Float16* W1h  = (_Float16*)(ws + 40 * MB);  // 8MB
  _Float16* W2h  = (_Float16*)(ws + 48 * MB);  // 8MB
  _Float16* z2   = (_Float16*)(ws + 56 * MB);  // 8MB  [B*S, D]
  float*    x1   = (float*)   (ws + 64 * MB);  // 16MB [B*S, D] f32
  _Float16* y2   = (_Float16*)(ws + 80 * MB);  // 8MB  (total 88MB)

  float* xout = (float*)d_out;
  float* att  = xout + (long)Bq * Sq * Dq;

  // Weight conversions (once per call; cheap vs. the GEMMs)
  auto cvt = [&](const float* in, _Float16* out, long n) {
    cvt_f16_kernel<<<dim3((unsigned)((n + 255) / 256)), 256, 0, stream>>>(in, out, n);
  };
  cvt(W_Q, Wq16, (long)Hq * DHq * Dq);
  cvt(W_K, Wk16, (long)Hq * DHq * Dq);
  cvt(W_V, Wv16, (long)Hq * DHq * Dq);
  cvt(w1,  W1h,  (long)Fq * Dq);
  cvt(w2,  W2h,  (long)Dq * Fq);
  cvt_wot_kernel<<<dim3((Dq * DHq * Hq + 255) / 256), 256, 0, stream>>>(W_O, WOt);

  // LN1
  ln_kernel<<<Bq * Sq, 256, 0, stream>>>(x, lnaw, lnab, yln);

  // QKV projections: per (b,h): [S x D] * [DH x D]^T   (M=2048 -> 16 tiles)
  gemm_nt_kernel<<<dim3(DHq / 64, Sq / 128, Bq * Hq), 128, 0, stream>>>(
      yln, (long)Sq * Dq, Hq, Dq, Wq16, (long)DHq * Dq, Hq, Dq,
      Qh, (long)Sq * DHq, DHq, nullptr, nullptr, Dq, /*f16*/ 2);
  gemm_nt_kernel<<<dim3(DHq / 64, Sq / 128, Bq * Hq), 128, 0, stream>>>(
      yln, (long)Sq * Dq, Hq, Dq, Wk16, (long)DHq * Dq, Hq, Dq,
      Kh, (long)Sq * DHq, DHq, nullptr, nullptr, Dq, 2);
  gemm_nt_kernel<<<dim3(DHq / 64, Sq / 128, Bq * Hq), 128, 0, stream>>>(
      yln, (long)Sq * Dq, Hq, Dq, Wv16, (long)DHq * Dq, Hq, Dq,
      Vth, (long)DHq * Sq, Sq, nullptr, nullptr, Dq, /*f16+transposed*/ 6);

  // Causal attention (writes normalized att f32 to d_out + z2 f16)
  attn_kernel<<<dim3(Sq / 64, Bq * Hq), 128, 0, stream>>>(Qh, Kh, Vth, att, z2);
  zero_att_kernel<<<dim3(Sq, Bq * Hq), 256, 0, stream>>>(att);

  // Output projection + residual: x1 = x + z2 @ WOt^T
  gemm_nt_kernel<<<dim3(Dq / 64, (Bq * Sq) / 128, 1), 128, 0, stream>>>(
      z2, 0L, 1, Dq, WOt, 0L, 1, Dq,
      x1, 0L, Dq, nullptr, x, Dq, /*f32*/ 0);

  // LN2
  ln_kernel<<<Bq * Sq, 256, 0, stream>>>(x1, lnmw, lnmb, y2);

  // MLP1: h = relu(y2 @ w1^T + b1)   (f16 output, overlays dead buffers)
  gemm_nt_kernel<<<dim3(Fq / 64, (Bq * Sq) / 128, 1), 128, 0, stream>>>(
      y2, 0L, 1, Dq, W1h, 0L, 1, Dq,
      h16, 0L, Fq, b1, nullptr, Dq, /*relu|f16*/ 3);

  // MLP2: out = x1 + h @ w2^T + b2   (final f32 x output)
  gemm_nt_kernel<<<dim3(Dq / 64, (Bq * Sq) / 128, 1), 128, 0, stream>>>(
      h16, 0L, 1, Fq, W2h, 0L, 1, Fq,
      xout, 0L, Dq, b2, x1, Fq, 0);
}
